// MultiHeadAttention_16466904613735
// MI455X (gfx1250) — matrix operand order repaired and checked by reading, not compiled, not run on hardware
//
#include <hip/hip_runtime.h>
#include <hip/hip_bf16.h>

// MI455X / gfx1250 multi-head attention, WMMA f16 pipeline.
// All matrix math via v_wmma_f32_16x16x32_f16 (wave32 fragments).

typedef __attribute__((ext_vector_type(16))) _Float16 v16h;
typedef __attribute__((ext_vector_type(8)))  float    v8f;

#define HH   12
#define NN   2048
#define DIN  768
#define DH   64
#define BB   4
#define MTOK (BB * NN)   // 8192 tokens

union AFrag { v16h h; unsigned int u[8]; };

// A-fragment (16x32 f16, MxK): lane m = lane&15, hi = lane>>4.
// VGPR r, half p -> K = (r<4?0:16) + hi*8 + (r&3)*2 + p  (pairs contiguous)
__device__ __forceinline__ int a_koff(int r, int hi) {
  return ((r & 4) << 2) + hi * 8 + ((r & 3) << 1);
}
// B-fragment (32x16 f16, KxN): lane n = lane&15, hi = lane>>4.
// VGPR r, half p -> K = (r<4?0:16) + hi*8 + (r&3)*2 + p  (pairs contiguous)
__device__ __forceinline__ int b_koff(int r, int hi) {
  return ((r & 4) << 2) + hi * 8 + ((r & 3) << 1);
}

__device__ __forceinline__ v8f wmma_f16(v16h a, v16h b, v8f c) {
  return __builtin_amdgcn_wmma_f32_16x16x32_f16(false, a, false, b, (short)0, c,
                                                false, false);
}

// ---------------------------------------------------------------- converts
__global__ void __launch_bounds__(256)
cvt_f32_to_f16(const float* __restrict__ src, _Float16* __restrict__ dst, int n) {
  int i = blockIdx.x * blockDim.x + threadIdx.x;
  if (i < n) dst[i] = (_Float16)src[i];
}

// W: [H][DIN][DH] f32 -> T: [H][DH][DIN] f16 (n-major for B-fragments)
__global__ void __launch_bounds__(256)
cvt_head_w(const float* __restrict__ W, _Float16* __restrict__ T) {
  int i = blockIdx.x * blockDim.x + threadIdx.x;
  if (i >= HH * DIN * DH) return;
  int d = i % DH;
  int k = (i / DH) % DIN;
  int h = i / (DH * DIN);
  T[((size_t)h * DH + d) * DIN + k] = (_Float16)W[i];
}

// Wo: [DIN][DIN] f32 -> WoT[n][k] = Wo[k][n] f16
__global__ void __launch_bounds__(256)
cvt_wo(const float* __restrict__ W, _Float16* __restrict__ T) {
  int i = blockIdx.x * blockDim.x + threadIdx.x;
  if (i >= DIN * DIN) return;
  int n = i % DIN;
  int k = i / DIN;
  T[(size_t)n * DIN + k] = (_Float16)W[i];
}

// --------------------------------------------------- per-head QKV projection
// One wave computes a 16x64 strip (all 4 n-tiles of the head): the A-fragment
// is loaded once per K-chunk and reused for 4 WMMAs.
// mode 0: Q (scaled by 1/8), 1: K, 2: V stored transposed.
__global__ void __launch_bounds__(128)
qkv_gemm_kernel(const _Float16* __restrict__ xh,   // [MTOK][DIN]
                const _Float16* __restrict__ wT,   // [H][DH][DIN]
                const float*    __restrict__ bias, // [H][DH]
                _Float16*       __restrict__ out, int mode) {
  const int lane = threadIdx.x & 31;
  const int wave = threadIdx.x >> 5;
  const int hi = lane >> 4, lo = lane & 15;
  const int h = blockIdx.y;
  const int mtile = blockIdx.x * 4 + wave;   // 0..511
  const int m0 = mtile * 16;

  const _Float16* xa = xh + (size_t)(m0 + lo) * DIN;
  const _Float16* wb = wT + (size_t)h * DH * DIN + (size_t)lo * DIN;

  v8f c0 = {}, c1 = {}, c2 = {}, c3 = {};
  for (int kc = 0; kc < DIN; kc += 32) {
    __builtin_prefetch(xa + kc + 64, 0, 0);  // global_prefetch_b8 (speculative)
    AFrag a;
#pragma unroll
    for (int r = 0; r < 8; ++r)
      a.u[r] = *(const unsigned int*)(xa + kc + a_koff(r, hi));
#pragma unroll
    for (int nt = 0; nt < 4; ++nt) {
      AFrag b;
#pragma unroll
      for (int r = 0; r < 8; ++r)
        b.u[r] = *(const unsigned int*)(wb + (size_t)(nt * 16) * DIN + kc +
                                        b_koff(r, hi));
      if      (nt == 0) c0 = wmma_f16(a.h, b.h, c0);
      else if (nt == 1) c1 = wmma_f16(a.h, b.h, c1);
      else if (nt == 2) c2 = wmma_f16(a.h, b.h, c2);
      else              c3 = wmma_f16(a.h, b.h, c3);
    }
  }

#pragma unroll
  for (int nt = 0; nt < 4; ++nt) {
    const int d = nt * 16 + lo;
    const float bvv = bias[h * DH + d];
    v8f& c = nt == 0 ? c0 : nt == 1 ? c1 : nt == 2 ? c2 : c3;
#pragma unroll
    for (int g = 0; g < 8; ++g) {
      int tokg = m0 + g + 8 * hi;
      float v = c[g] + bvv;
      if (mode == 0) v *= 0.125f;             // fold 1/sqrt(64) into Q
      _Float16 hv = (_Float16)v;
      int b_ = tokg >> 11, tok = tokg & (NN - 1);
      if (mode == 2)   // V transposed: [B][H][DH][NN]
        out[(((size_t)(b_ * HH + h)) * DH + d) * NN + tok] = hv;
      else             // Q/K: [B][H][NN][DH]
        out[(((size_t)(b_ * HH + h)) * NN + tok) * DH + d] = hv;
    }
  }
}

// ------------------------------------------------- fused flash attention
// One wave = 16 query rows; streams 32 keys/step; online softmax.
__global__ void __launch_bounds__(128)
attn_kernel(const _Float16* __restrict__ qh,  // [B][H][NN][DH], pre-scaled
            const _Float16* __restrict__ kh,  // [B][H][NN][DH]
            const _Float16* __restrict__ vt,  // [B][H][DH][NN]
            _Float16*       __restrict__ ctxh) { // [B][NN][DIN]
  __shared__ __align__(16) _Float16 Plds[4][16][32];
  const int lane = threadIdx.x & 31;
  const int wave = threadIdx.x >> 5;
  const int hi = lane >> 4, lo = lane & 15;
  const int bh = blockIdx.y;
  const int b = bh / HH, h = bh % HH;
  const int m0 = (blockIdx.x * 4 + wave) * 16;

  const _Float16* qp = qh + (size_t)bh * NN * DH;
  const _Float16* kp = kh + (size_t)bh * NN * DH;
  const _Float16* vp = vt + (size_t)bh * DH * NN;

  AFrag aq[2];
#pragma unroll
  for (int ch = 0; ch < 2; ++ch)
#pragma unroll
    for (int r = 0; r < 8; ++r)
      aq[ch].u[r] =
          *(const unsigned int*)(qp + (size_t)(m0 + lo) * DH + ch * 32 + a_koff(r, hi));

  v8f O0 = {}, O1 = {}, O2 = {}, O3 = {};
  float mrow[8], srow[8];
#pragma unroll
  for (int g = 0; g < 8; ++g) { mrow[g] = -3.0e38f; srow[g] = 0.0f; }

  for (int j = 0; j < NN; j += 32) {
    // ---- S = Q @ K^T for two 16-key sub-blocks
    v8f cs0 = {}, cs1 = {};
#pragma unroll
    for (int ch = 0; ch < 2; ++ch) {
      AFrag bk0, bk1;
#pragma unroll
      for (int r = 0; r < 8; ++r) {
        int k = ch * 32 + b_koff(r, hi);
        bk0.u[r] = *(const unsigned int*)(kp + (size_t)(j + lo) * DH + k);
        bk1.u[r] = *(const unsigned int*)(kp + (size_t)(j + 16 + lo) * DH + k);
      }
      cs0 = wmma_f16(aq[ch].h, bk0.h, cs0);
      cs1 = wmma_f16(aq[ch].h, bk1.h, cs1);
    }

    // ---- online softmax (row stats shared across 16-lane half-wave)
    float al[8];
#pragma unroll
    for (int g = 0; g < 8; ++g) {
      float v0 = cs0[g], v1 = cs1[g];
      float mx = fmaxf(v0, v1);
      mx = fmaxf(mx, __shfl_xor(mx, 1, 32));
      mx = fmaxf(mx, __shfl_xor(mx, 2, 32));
      mx = fmaxf(mx, __shfl_xor(mx, 4, 32));
      mx = fmaxf(mx, __shfl_xor(mx, 8, 32));
      float mnew = fmaxf(mrow[g], mx);
      float a = __expf(mrow[g] - mnew);
      mrow[g] = mnew;
      float p0 = __expf(v0 - mnew);
      float p1 = __expf(v1 - mnew);
      float rs = p0 + p1;
      rs += __shfl_xor(rs, 1, 32);
      rs += __shfl_xor(rs, 2, 32);
      rs += __shfl_xor(rs, 4, 32);
      rs += __shfl_xor(rs, 8, 32);
      srow[g] = srow[g] * a + rs;
      al[g] = a;
      int M = g + 8 * hi;                 // C-fragment row this lane holds
      Plds[wave][M][lo]      = (_Float16)p0;
      Plds[wave][M][16 + lo] = (_Float16)p1;
    }
#pragma unroll
    for (int g = 0; g < 8; ++g) {
      O0[g] *= al[g]; O1[g] *= al[g]; O2[g] *= al[g]; O3[g] *= al[g];
    }

    // LDS ops are in-order per wave; wait for P stores, then gather A-fragment.
    __builtin_amdgcn_wave_barrier();
    asm volatile("s_wait_dscnt 0x0" ::: "memory");
    AFrag ap;
#pragma unroll
    for (int r = 0; r < 8; ++r)
      ap.u[r] = *(const unsigned int*)(&Plds[wave][lo][a_koff(r, hi)]);
    __builtin_amdgcn_wave_barrier();

    // ---- O += P @ V  (V transposed -> contiguous key pairs per lane)
#pragma unroll
    for (int nb = 0; nb < 4; ++nb) {
      AFrag bvf;
#pragma unroll
      for (int r = 0; r < 8; ++r)
        bvf.u[r] =
            *(const unsigned int*)(vp + (size_t)(nb * 16 + lo) * NN + j + b_koff(r, hi));
      if      (nb == 0) O0 = wmma_f16(ap.h, bvf.h, O0);
      else if (nb == 1) O1 = wmma_f16(ap.h, bvf.h, O1);
      else if (nb == 2) O2 = wmma_f16(ap.h, bvf.h, O2);
      else              O3 = wmma_f16(ap.h, bvf.h, O3);
    }
  }

#pragma unroll
  for (int g = 0; g < 8; ++g) {
    int tok = m0 + g + 8 * hi;
    float inv = 1.0f / srow[g];
    size_t base = ((size_t)(b * NN + tok)) * DIN + h * DH;
    ctxh[base + 0 * 16 + lo] = (_Float16)(O0[g] * inv);
    ctxh[base + 1 * 16 + lo] = (_Float16)(O1[g] * inv);
    ctxh[base + 2 * 16 + lo] = (_Float16)(O2[g] * inv);
    ctxh[base + 3 * 16 + lo] = (_Float16)(O3[g] * inv);
  }
}

// ------------------------------------------------------- output projection
// One wave computes a 16x64 strip (4 accumulators, A-fragment reused 4x).
__global__ void __launch_bounds__(128)
out_gemm_kernel(const _Float16* __restrict__ ctxh, // [MTOK][DIN]
                const _Float16* __restrict__ woT,  // [DIN][DIN] n-major
                const float*    __restrict__ bo,
                float*          __restrict__ out) {
  const int lane = threadIdx.x & 31;
  const int wave = threadIdx.x >> 5;
  const int hi = lane >> 4, lo = lane & 15;
  const int n0 = blockIdx.x * 64;            // 0..11 supertiles of N
  const int mtile = blockIdx.y * 4 + wave;   // 0..511
  const int m0 = mtile * 16;

  const _Float16* xa = ctxh + (size_t)(m0 + lo) * DIN;
  const _Float16* wb = woT + (size_t)(n0 + lo) * DIN;

  v8f c0 = {}, c1 = {}, c2 = {}, c3 = {};
  for (int kc = 0; kc < DIN; kc += 32) {
    __builtin_prefetch(xa + kc + 64, 0, 0);
    AFrag a;
#pragma unroll
    for (int r = 0; r < 8; ++r)
      a.u[r] = *(const unsigned int*)(xa + kc + a_koff(r, hi));
#pragma unroll
    for (int nt = 0; nt < 4; ++nt) {
      AFrag b;
#pragma unroll
      for (int r = 0; r < 8; ++r)
        b.u[r] = *(const unsigned int*)(wb + (size_t)(nt * 16) * DIN + kc +
                                        b_koff(r, hi));
      if      (nt == 0) c0 = wmma_f16(a.h, b.h, c0);
      else if (nt == 1) c1 = wmma_f16(a.h, b.h, c1);
      else if (nt == 2) c2 = wmma_f16(a.h, b.h, c2);
      else              c3 = wmma_f16(a.h, b.h, c3);
    }
  }

#pragma unroll
  for (int nt = 0; nt < 4; ++nt) {
    const int d = n0 + nt * 16 + lo;
    const float bvv = bo[d];
    v8f& c = nt == 0 ? c0 : nt == 1 ? c1 : nt == 2 ? c2 : c3;
#pragma unroll
    for (int g = 0; g < 8; ++g) {
      int tokg = m0 + g + 8 * hi;
      out[(size_t)tokg * DIN + d] = c[g] + bvv;
    }
  }
}

// ------------------------------------------------------------------ launch
extern "C" void kernel_launch(void* const* d_in, const int* in_sizes, int n_in,
                              void* d_out, int out_size, void* d_ws, size_t ws_size,
                              hipStream_t stream) {
  const float* x  = (const float*)d_in[0];
  const float* Wq = (const float*)d_in[1];
  const float* bq = (const float*)d_in[2];
  const float* Wk = (const float*)d_in[3];
  const float* bk = (const float*)d_in[4];
  const float* Wv = (const float*)d_in[5];
  const float* bv = (const float*)d_in[6];
  const float* Wo = (const float*)d_in[7];
  const float* bo = (const float*)d_in[8];
  float* out = (float*)d_out;
  (void)in_sizes; (void)n_in; (void)out_size; (void)ws_size;

  // Workspace carve-out (~66 MB total), 256B aligned slices.
  char* ws = (char*)d_ws;
  size_t off = 0;
  auto take = [&](size_t elems) {
    _Float16* p = (_Float16*)(ws + off);
    off = (off + elems * sizeof(_Float16) + 255) & ~(size_t)255;
    return p;
  };
  _Float16* xh   = take((size_t)MTOK * DIN);
  _Float16* wqT  = take((size_t)HH * DH * DIN);
  _Float16* wkT  = take((size_t)HH * DH * DIN);
  _Float16* wvT  = take((size_t)HH * DH * DIN);
  _Float16* woT  = take((size_t)DIN * DIN);
  _Float16* qh   = take((size_t)BB * HH * NN * DH);
  _Float16* kh   = take((size_t)BB * HH * NN * DH);
  _Float16* vt   = take((size_t)BB * HH * NN * DH);
  _Float16* ctxh = take((size_t)MTOK * DIN);

  { // x -> f16
    int n = MTOK * DIN;
    cvt_f32_to_f16<<<dim3((n + 255) / 256), dim3(256), 0, stream>>>(x, xh, n);
  }
  { // transposed f16 weights
    int nw = HH * DIN * DH;
    cvt_head_w<<<dim3((nw + 255) / 256), dim3(256), 0, stream>>>(Wq, wqT);
    cvt_head_w<<<dim3((nw + 255) / 256), dim3(256), 0, stream>>>(Wk, wkT);
    cvt_head_w<<<dim3((nw + 255) / 256), dim3(256), 0, stream>>>(Wv, wvT);
    int no = DIN * DIN;
    cvt_wo<<<dim3((no + 255) / 256), dim3(256), 0, stream>>>(Wo, woT);
  }

  dim3 gq(MTOK / 64, HH);            // (128, 12), 4 waves/block, 16x64 per wave
  qkv_gemm_kernel<<<gq, dim3(128), 0, stream>>>(xh, wqT, bq, qh, 0);
  qkv_gemm_kernel<<<gq, dim3(128), 0, stream>>>(xh, wkT, bk, kh, 1);
  qkv_gemm_kernel<<<gq, dim3(128), 0, stream>>>(xh, wvT, bv, vt, 2);

  attn_kernel<<<dim3(NN / 64, BB * HH), dim3(128), 0, stream>>>(qh, kh, vt, ctxh);

  out_gemm_kernel<<<dim3(DIN / 64, MTOK / 64), dim3(128), 0, stream>>>(ctxh, woT, bo, out);
}
